// Conv2d_34522947125875
// MI455X (gfx1250) — matrix operand order repaired and checked by reading, not compiled, not run on hardware
//
#include <hip/hip_runtime.h>

// Conv2d 256x256x3x3 on 256x256, PAD=1, STRIDE=1, fp32 in/out.
// Round 2: split-bf16 (bf16x3) WMMA implicit GEMM with:
//   - pre-pass kernels converting x/weights to bf16 hi/lo in d_ws
//     (x stored pixel-major + zero-padded border -> no edge predicates)
//   - main loop stages tiles with global_load_async_to_lds_b128 (ASYNCcnt),
//     double-buffered to overlap DMA with v_wmma_f32_16x16x32_bf16
// Workspace requirement: ~70.6 MB (2x padded-x bf16 + 2x weight bf16).

#define CIN   256
#define COUT  256
#define HW    256
#define KS    40           // LDS row stride (bf16 elems): 80B, 16B-aligned, conflict-spread
#define PADW  258          // padded pixel grid 258x258
#define XT_ELEMS ((size_t)PADW * PADW * CIN)   // 17,040,384
#define W_ELEMS  ((size_t)9 * COUT * CIN)      //    589,824

typedef __attribute__((ext_vector_type(16))) __bf16 v16bf;
typedef __attribute__((ext_vector_type(8)))  float  v8f;

union Frag { v16bf v; uint4 q[2]; };

__device__ __forceinline__ unsigned short f2bf_rn(float f) {
  unsigned u = __float_as_uint(f);
  u += 0x7FFFu + ((u >> 16) & 1u);          // round-to-nearest-even to bf16
  return (unsigned short)(u >> 16);
}

// ---------------------------------------------------------------------------
// Pre-pass 1: zero the padded border rows of xt (hi & lo).
// 1028 border pixels x 256 ci x 2B = 512B per pixel = 32 uint4 chunks.
__global__ __launch_bounds__(256)
void zero_pad_kernel(unsigned short* __restrict__ xh, unsigned short* __restrict__ xl) {
  const int i = blockIdx.x * 256 + threadIdx.x;
  if (i >= 1028 * 32) return;
  const int rowi = i >> 5, c = i & 31;
  size_t p;
  if (rowi < 258)      p = (size_t)rowi;                          // top row
  else if (rowi < 516) p = (size_t)257 * PADW + (rowi - 258);     // bottom row
  else if (rowi < 772) p = (size_t)(rowi - 515) * PADW;           // left col, rows 1..256
  else                 p = (size_t)(rowi - 771) * PADW + 257;     // right col, rows 1..256
  const uint4 z = {0u, 0u, 0u, 0u};
  *(uint4*)&xh[p * CIN + c * 8] = z;
  *(uint4*)&xl[p * CIN + c * 8] = z;
}

// ---------------------------------------------------------------------------
// Pre-pass 2: weights (oc,ci,kh,kw) fp32 -> wt[tap][oc][ci] bf16 hi/lo.
__global__ __launch_bounds__(256)
void convert_w_kernel(const float* __restrict__ wgt,
                      unsigned short* __restrict__ wh, unsigned short* __restrict__ wl) {
  const int i = blockIdx.x * 256 + threadIdx.x;
  if (i >= (int)W_ELEMS) return;
  const int tap = i >> 16, rem = i & 65535, oc = rem >> 8, ci = rem & 255;
  const float f = wgt[(size_t)oc * (CIN * 9) + ci * 9 + tap];
  const unsigned short h16 = f2bf_rn(f);
  wh[i] = h16;
  wl[i] = f2bf_rn(f - __uint_as_float((unsigned)h16 << 16));
}

// ---------------------------------------------------------------------------
// Pre-pass 3: x (ci,h,w) fp32 -> xt[(h+1)*258 + (w+1)][ci] bf16 hi/lo.
// 64ci x 64w tile through LDS: coalesced reads along w, 128B row writes.
__global__ __launch_bounds__(256)
void convert_x_kernel(const float* __restrict__ x,
                      unsigned short* __restrict__ xh, unsigned short* __restrict__ xl) {
  __shared__ alignas(16) unsigned short th[64 * 72];
  __shared__ alignas(16) unsigned short tl[64 * 72];
  const int tid = threadIdx.x;
  const int ci0 = blockIdx.y << 6;
  const int hrow = blockIdx.x >> 2;
  const int wbase = (blockIdx.x & 3) << 6;

  for (int i = tid; i < 64 * 64; i += 256) {
    const int wo = i & 63, co = i >> 6;                 // lanes -> consecutive w
    const float f = x[(size_t)(ci0 + co) * (HW * HW) + hrow * HW + wbase + wo];
    const unsigned short h16 = f2bf_rn(f);
    th[wo * 72 + co] = h16;
    tl[wo * 72 + co] = f2bf_rn(f - __uint_as_float((unsigned)h16 << 16));
  }
  __syncthreads();
  #pragma unroll
  for (int s = 0; s < 2; ++s) {
    const int idx = tid + (s << 8);
    const int px = idx >> 3, c8 = idx & 7;              // 8x 16B chunks per 128B row
    const size_t p = (size_t)(hrow + 1) * PADW + (wbase + px + 1);
    const size_t dst = p * CIN + ci0 + c8 * 8;
    *(uint4*)&xh[dst] = *(const uint4*)&th[px * 72 + c8 * 8];
    *(uint4*)&xl[dst] = *(const uint4*)&tl[px * 72 + c8 * 8];
  }
}

// ---------------------------------------------------------------------------
// Main kernel: 128 oc x 128 px per block, 8 waves, async double-buffered LDS.
#define ASYNC_B128(ldsptr, goff, base)                                         \
  asm volatile("global_load_async_to_lds_b128 %0, %1, %2"                      \
               :: "v"((unsigned)(size_t)(ldsptr)), "v"(goff),                  \
                  "s"((unsigned long long)(base)) : "memory")

__global__ __launch_bounds__(256)
void conv2d_wmma_async(const unsigned short* __restrict__ xh,
                       const unsigned short* __restrict__ xl,
                       const unsigned short* __restrict__ wh,
                       const unsigned short* __restrict__ wl,
                       const float* __restrict__ bias,
                       float* __restrict__ out) {
  __shared__ alignas(16) unsigned short sAh[2][128 * KS];   // 128 oc x 32 ci
  __shared__ alignas(16) unsigned short sAl[2][128 * KS];
  __shared__ alignas(16) unsigned short sBh[2][128 * KS];   // 128 px x 32 ci
  __shared__ alignas(16) unsigned short sBl[2][128 * KS];

  const int tid  = threadIdx.x;
  const int lane = tid & 31, wid = tid >> 5;
  const int hrow  = blockIdx.x >> 1;
  const int wbase = (blockIdx.x & 1) << 7;
  const int ocbase = blockIdx.y << 7;

  // stage reduction chunk `it` (it = kh*24 + kw*8 + cc) into buffer `buf`
  auto stage = [&](int it, int buf) {
    const int cc  = it & 7;
    const int kwv = (it >> 3) % 3;
    const int khv = it / 24;
    const int tap = khv * 3 + kwv;
    // A: 128 oc x 32 ci, 16B per async op, 2 ops/thread/array
    #pragma unroll
    for (int s = 0; s < 2; ++s) {
      const int idx = tid + (s << 8);
      const int oco = idx >> 2, c16 = idx & 3;
      const unsigned goff =
          ((unsigned)(tap * 65536 + (ocbase + oco) * 256 + (cc << 5)) << 1) + (c16 << 4);
      ASYNC_B128(&sAh[buf][oco * KS + (c16 << 3)], goff, wh);
      ASYNC_B128(&sAl[buf][oco * KS + (c16 << 3)], goff, wl);
    }
    // B: 128 px x 32 ci from padded pixel-major x; no predicates needed
    const int prow = (hrow + khv) * PADW + wbase + kwv;   // padded row for px=0
    #pragma unroll
    for (int s = 0; s < 2; ++s) {
      const int idx = tid + (s << 8);
      const int px = idx >> 2, c16 = idx & 3;
      const unsigned goff =
          ((unsigned)((prow + px) * 256 + (cc << 5)) << 1) + (c16 << 4);
      ASYNC_B128(&sBh[buf][px * KS + (c16 << 3)], goff, xh);
      ASYNC_B128(&sBl[buf][px * KS + (c16 << 3)], goff, xl);
    }
  };

  v8f acc[8] = {};
  const int r = lane & 15, half = lane >> 4;

  stage(0, 0);
  asm volatile("s_wait_asynccnt 0" ::: "memory");
  __syncthreads();

  for (int it = 0; it < 72; ++it) {
    const int buf = it & 1;
    if (it < 71) stage(it + 1, buf ^ 1);   // overlap DMA with WMMA

    // A fragments: lanes 0-15: M=r, K{0..7,16..23}; lanes 16-31: M=r, K{8..15,24..31}
    Frag ah, al;
    {
      const unsigned short* pa = &sAh[buf][(wid * 16 + r) * KS + half * 8];
      ah.q[0] = *(const uint4*)(pa);
      ah.q[1] = *(const uint4*)(pa + 16);
      const unsigned short* pl = &sAl[buf][(wid * 16 + r) * KS + half * 8];
      al.q[0] = *(const uint4*)(pl);
      al.q[1] = *(const uint4*)(pl + 16);
    }
    #pragma unroll
    for (int n = 0; n < 8; ++n) {
      Frag bh, bl;
      const unsigned short* pb = &sBh[buf][(n * 16 + r) * KS + half * 8];
      bh.q[0] = *(const uint4*)(pb);
      bh.q[1] = *(const uint4*)(pb + 16);
      const unsigned short* pc = &sBl[buf][(n * 16 + r) * KS + half * 8];
      bl.q[0] = *(const uint4*)(pc);
      bl.q[1] = *(const uint4*)(pc + 16);

      acc[n] = __builtin_amdgcn_wmma_f32_16x16x32_bf16(
                 false, ah.v, false, bh.v, (short)0, acc[n], false, false);
      acc[n] = __builtin_amdgcn_wmma_f32_16x16x32_bf16(
                 false, al.v, false, bh.v, (short)0, acc[n], false, false);
      acc[n] = __builtin_amdgcn_wmma_f32_16x16x32_bf16(
                 false, ah.v, false, bl.v, (short)0, acc[n], false, false);
    }

    asm volatile("s_wait_asynccnt 0" ::: "memory");
    __syncthreads();
  }

  // epilogue: C layout (VGPR j: M = j + 8*half, N = r), add bias
  #pragma unroll
  for (int n = 0; n < 8; ++n) {
    #pragma unroll
    for (int j = 0; j < 8; ++j) {
      const int oc   = ocbase + wid * 16 + j + half * 8;
      const int wcol = wbase + n * 16 + r;
      out[(size_t)oc * (HW * HW) + hrow * HW + wcol] = acc[n][j] + bias[oc];
    }
  }
}

// ---------------------------------------------------------------------------
extern "C" void kernel_launch(void* const* d_in, const int* in_sizes, int n_in,
                              void* d_out, int out_size, void* d_ws, size_t ws_size,
                              hipStream_t stream) {
  const float* x    = (const float*)d_in[0];   // (256, 256, 256)
  const float* wgt  = (const float*)d_in[1];   // (256, 256, 3, 3)
  const float* bias = (const float*)d_in[2];   // (256, 1, 1)
  float* out = (float*)d_out;                  // (256, 256, 256)

  // workspace layout (bf16): xh | xl | wh | wl  (~70.5 MB total)
  unsigned short* xh = (unsigned short*)d_ws;
  unsigned short* xl = xh + XT_ELEMS;
  unsigned short* wh = xl + XT_ELEMS;
  unsigned short* wl = wh + W_ELEMS;

  zero_pad_kernel<<<(1028 * 32 + 255) / 256, 256, 0, stream>>>(xh, xl);
  convert_w_kernel<<<((int)W_ELEMS + 255) / 256, 256, 0, stream>>>(wgt, wh, wl);
  convert_x_kernel<<<dim3(1024, 4), 256, 0, stream>>>(x, xh, xl);
  conv2d_wmma_async<<<dim3(512, 2), 256, 0, stream>>>(xh, xl, wh, wl, bias, out);
}
